// GPT2_46694884442567
// MI455X (gfx1250) — matrix operand order repaired and checked by reading, not compiled, not run on hardware
//
#include <hip/hip_runtime.h>
#include <cstddef>

typedef __attribute__((ext_vector_type(16))) __bf16 bf16x16;
typedef __attribute__((ext_vector_type(8)))  __bf16 bf16x8;
typedef __attribute__((ext_vector_type(8)))  float  f32x8;
typedef __attribute__((ext_vector_type(4)))  unsigned int u32x4;
typedef __attribute__((ext_vector_type(4)))  int    i32x4;
typedef __attribute__((ext_vector_type(8)))  int    i32x8;

static constexpr int kL = 11, kH = 12, kE = 768, kV = 50257;
static constexpr int kB = 2, kT = 1024, kD = 64;
static constexpr int kTok = kB * kT;   // 2048 tokens

#if __has_builtin(__builtin_amdgcn_tensor_load_to_lds)
#define HAVE_TDM 1
#endif

// ---------------------------------------------------------------------------
// CDNA5 async copy: global -> LDS, 16B per lane, tracked by ASYNCcnt.
// lds_off = low 32 bits of generic pointer to __shared__ (== LDS byte offset).
// ---------------------------------------------------------------------------
__device__ __forceinline__ void async_ld16(unsigned lds_off, const void* gaddr) {
  asm volatile("global_load_async_to_lds_b128 %0, %1, off"
               :: "v"(lds_off), "v"(gaddr) : "memory");
}
__device__ __forceinline__ void wait_async0() {
  asm volatile("s_wait_asynccnt 0x0" ::: "memory");
}

// ---------------------------------------------------------------------------
// Tensor Data Mover: one instruction DMAs a 64x64 bf16 tile (global, row
// stride `row_stride_elems`) into LDS with a 16B pad after each 128B row
// (-> 144B LDS row stride, matching the padded tile layout below).
// D# packing per CDNA5 ISA 8.3/8.4. Issue from one wave only; TENSORcnt.
// ---------------------------------------------------------------------------
__device__ __forceinline__ void tdm_load_tile_64x64_bf16(unsigned lds_addr,
                                                         const void* gaddr,
                                                         unsigned row_stride_elems) {
#if defined(HAVE_TDM)
  unsigned long long ga = (unsigned long long)gaddr;
  u32x4 g0;
  g0[0] = 1u;                                   // count=1 (valid user D#)
  g0[1] = lds_addr;                             // lds_addr [63:32]
  g0[2] = (unsigned)ga;                         // global_addr[31:0]
  g0[3] = (unsigned)((ga >> 32) & 0x01FFFFFFu)  // global_addr[56:32]
        | 0x80000000u;                          // type=2 ("image")
  i32x8 g1;
  g1[0] = (1 << 16)                             // data_size = 2 bytes
        | (1 << 20)                             // pad_enable
        | (4 << 22)                             // pad_interval: 32 DWORDs (128B row)
        | (3 << 25);                            // pad_amount: 4 DWORDs (16B)
  g1[1] = 0;                                    // tensor_dim0[15:0] = 0 (dim0 = 1<<30)
  g1[2] = 0x4000;                               // tensor_dim0[31:16]; dim1 low = 0
  g1[3] = 0x4000 | (64 << 16);                  // tensor_dim1[31:16]; tile_dim0 = 64
  g1[4] = 64;                                   // tile_dim1 = 64; tile_dim2 = 0
  g1[5] = (int)row_stride_elems;                // tensor_dim0_stride[31:0]
  g1[6] = 0;
  g1[7] = 0;
  i32x4 gz = {0, 0, 0, 0};
#if defined(__clang_major__) && __clang_major__ >= 23
  i32x8 gz8 = {0, 0, 0, 0, 0, 0, 0, 0};
  __builtin_amdgcn_tensor_load_to_lds(g0, g1, gz, gz, gz8, 0);
#else
  __builtin_amdgcn_tensor_load_to_lds(g0, g1, gz, gz, 0);
#endif
#else
  (void)lds_addr; (void)gaddr; (void)row_stride_elems;
#endif
}

// ---------------------------------------------------------------------------
// WMMA fragment helpers (CDNA5 16x16x32 bf16, wave32)
// A-frag (16x32, M=lane%16): elems 0..7 -> K = base+0..7, 8..15 -> K = base+16..23,
//   base = (lane<16) ? 0 : 8.   Two 16B LDS loads per fragment.
// B-frag (32x16, N=lane%16): elems 0..7 -> K = klo/2+0..7, 8..15 -> K = klo/2+16..23, klo = (lane<16)?0:16.
// C/D-frag: elem r at lane L = C[r + (L>=16?8:0)][L%16].
// ---------------------------------------------------------------------------
__device__ __forceinline__ bf16x16 frag_a(const __bf16* row, int abase) {
  bf16x8 lo = *(const bf16x8*)(row + abase);
  bf16x8 hi = *(const bf16x8*)(row + abase + 16);
  bf16x16 r;
#pragma unroll
  for (int i = 0; i < 8; ++i) { r[i] = lo[i]; r[i + 8] = hi[i]; }
  return r;
}

__device__ __forceinline__ bf16x16 frag_b(const __bf16* row, int klo) {
  bf16x8 lo = *(const bf16x8*)(row + klo / 2);
  bf16x8 hi = *(const bf16x8*)(row + klo / 2 + 16);
  bf16x16 r;
#pragma unroll
  for (int i = 0; i < 8; ++i) { r[i] = lo[i]; r[i + 8] = hi[i]; }
  return r;
}

__device__ __forceinline__ f32x8 wmma_bf16(bf16x16 a, bf16x16 b, f32x8 c) {
  return __builtin_amdgcn_wmma_f32_16x16x32_bf16(false, a, false, b, (short)0, c,
                                                 false, false);
}

// ---------------------------------------------------------------------------
// Token embedding: x[tok] = wte[ids[tok]] + wpe[tok % T]   (fp32 residual)
// ---------------------------------------------------------------------------
__global__ void embed_kernel(const int* __restrict__ ids, const float* __restrict__ wte,
                             const float* __restrict__ wpe, float* __restrict__ x) {
  int tok = blockIdx.x;
  int t = tok % kT;
  int id = ids[tok];
#pragma unroll
  for (int i = 0; i < 3; ++i) {
    int e = threadIdx.x + i * 256;
    x[(size_t)tok * kE + e] = wte[(size_t)id * kE + e] + wpe[(size_t)t * kE + e];
  }
}

// ---------------------------------------------------------------------------
// LayerNorm over E=768, one block (256 thr) per token, bf16 output
// ---------------------------------------------------------------------------
__global__ void ln_kernel(const float* __restrict__ x, const float* __restrict__ w,
                          const float* __restrict__ b, __bf16* __restrict__ out) {
  __shared__ float red[256];
  int tok = blockIdx.x, tid = threadIdx.x;
  float v[3];
  float s = 0.f;
#pragma unroll
  for (int i = 0; i < 3; ++i) { v[i] = x[(size_t)tok * kE + tid + i * 256]; s += v[i]; }
  red[tid] = s; __syncthreads();
  for (int st = 128; st > 0; st >>= 1) { if (tid < st) red[tid] += red[tid + st]; __syncthreads(); }
  float mean = red[0] * (1.0f / kE); __syncthreads();
  float ss = 0.f;
#pragma unroll
  for (int i = 0; i < 3; ++i) { float d = v[i] - mean; ss += d * d; }
  red[tid] = ss; __syncthreads();
  for (int st = 128; st > 0; st >>= 1) { if (tid < st) red[tid] += red[tid + st]; __syncthreads(); }
  float rstd = rsqrtf(red[0] * (1.0f / kE) + 1e-5f);
#pragma unroll
  for (int i = 0; i < 3; ++i) {
    int e = tid + i * 256;
    out[(size_t)tok * kE + e] = (__bf16)((v[i] - mean) * rstd * w[e] + b[e]);
  }
}

// ---------------------------------------------------------------------------
// WMMA GEMM: C[M=2048, N] = A(bf16)[M,K] * W(fp32)[K,N] (+bias) (+epilogue)
// Block: 64x64 tile, 128 threads = 4 waves, wave -> 32x32 (2x2 wmma tiles).
// Grid: (m-tiles in x, n-tiles in y) -> consecutive blocks share the same
// weight column-block for L2 reuse (weights dominate HBM traffic).
// MODE 0: out bf16 = acc + bias                        (QKV)
// MODE 1: out fp32 = resid + acc + bias                (attn-proj / MLP-proj)
// MODE 2: out bf16 = gelu_tanh(acc + bias)             (FC)
// MODE 3: out fp32 = acc, N-guarded                    (LM head)
// ---------------------------------------------------------------------------
template <int MODE>
__global__ void gemm_kernel(const __bf16* __restrict__ A, const float* __restrict__ W,
                            const float* __restrict__ bias, float* __restrict__ resid,
                            float* __restrict__ outf, __bf16* __restrict__ outb,
                            int K, int N) {
  __shared__ __bf16 As[64 * 40];   // [m][k], row stride 40 halves (80B, 16B aligned)
  __shared__ __bf16 Bs[64 * 40];   // [n][k] (transposed weights)
  const int tid = threadIdx.x;
  const int lane = tid & 31;
  const int wid = tid >> 5;
  const int mw = wid >> 1, nw = wid & 1;
  const int mrow = lane & 15;
  const int abase = (lane < 16) ? 0 : 8;
  const int klo = (lane < 16) ? 0 : 16;
  const int m0 = blockIdx.x * 64;
  const int n0 = blockIdx.y * 64;

  const f32x8 vzero = {0.f, 0.f, 0.f, 0.f, 0.f, 0.f, 0.f, 0.f};
  f32x8 acc[2][2];
#pragma unroll
  for (int mt = 0; mt < 2; ++mt)
#pragma unroll
    for (int nt = 0; nt < 2; ++nt) acc[mt][nt] = vzero;

  for (int k0 = 0; k0 < K; k0 += 32) {
    // A tile (bf16, pure copy) via async global->LDS, 16B per lane
#pragma unroll
    for (int c = 0; c < 2; ++c) {
      int chunk = tid + c * 128;
      int row = chunk >> 2, cc = chunk & 3;
      async_ld16((unsigned)(size_t)&As[row * 40 + cc * 8],
                 &A[(size_t)(m0 + row) * K + k0 + cc * 8]);
    }
    // W tile: fp32 [k][n] -> bf16 [n][k] (needs VALU convert -> normal path)
#pragma unroll
    for (int g = 0; g < 4; ++g) {
      int gid = tid + g * 128;           // 0..511
      int n = gid & 63, kg = gid >> 6;   // kg: 0..7
      int gn = n0 + n;
      __bf16 tmp[4];
#pragma unroll
      for (int i = 0; i < 4; ++i) {
        int k = k0 + kg * 4 + i;
        float wv = (MODE != 3 || gn < N) ? W[(size_t)k * N + gn] : 0.f;
        tmp[i] = (__bf16)wv;
      }
      *(uint2*)&Bs[n * 40 + kg * 4] = *(const uint2*)tmp;
    }
    if (k0 + 32 < K)  // global_prefetch_b8 of next weight K-tile
      __builtin_prefetch(&W[(size_t)(k0 + 32) * N + n0 + (tid & 63)], 0, 0);
    wait_async0();
    __syncthreads();

    bf16x16 af[2], bfr[2];
#pragma unroll
    for (int mt = 0; mt < 2; ++mt)
      af[mt] = frag_a(&As[(mw * 32 + mt * 16 + mrow) * 40], abase);
#pragma unroll
    for (int nt = 0; nt < 2; ++nt)
      bfr[nt] = frag_b(&Bs[(nw * 32 + nt * 16 + mrow) * 40], klo);
#pragma unroll
    for (int mt = 0; mt < 2; ++mt)
#pragma unroll
      for (int nt = 0; nt < 2; ++nt)
        acc[mt][nt] = wmma_bf16(af[mt], bfr[nt], acc[mt][nt]);
    __syncthreads();
  }

  const int h8 = (lane >= 16) ? 8 : 0;
  const int ncol = lane & 15;
#pragma unroll
  for (int mt = 0; mt < 2; ++mt) {
#pragma unroll
    for (int nt = 0; nt < 2; ++nt) {
#pragma unroll
      for (int r = 0; r < 8; ++r) {
        int row = m0 + mw * 32 + mt * 16 + r + h8;
        int col = n0 + nw * 32 + nt * 16 + ncol;
        float v = acc[mt][nt][r];
        if (MODE != 3) v += bias[col];
        size_t idx = (size_t)row * N + col;
        if (MODE == 0) {
          outb[idx] = (__bf16)v;
        } else if (MODE == 1) {
          outf[idx] = resid[idx] + v;
        } else if (MODE == 2) {
          float u = v + 0.044715f * v * v * v;
          float g = 0.5f * v * (1.f + tanhf(0.7978845608f * u));
          outb[idx] = (__bf16)g;
        } else {
          if (col < N) outf[idx] = v;
        }
      }
    }
  }
}

// ---------------------------------------------------------------------------
// Flash attention, causal. One block (128 thr, 4 waves) per (q-tile, head, batch).
// Wave w owns q rows [w*16, w*16+16). S = Q K^T and O += P V via WMMA.
// Staging: Q via async global->LDS, K via TDM tensor_load_to_lds (wave 0,
// TENSORcnt), V via manual transpose. Q frags re-read from resident Qs.
// ---------------------------------------------------------------------------
__global__ void attn_kernel(const __bf16* __restrict__ qkv, __bf16* __restrict__ y) {
  __shared__ __bf16 Qs[64 * 72];   // [q][d]   row stride 72 halves (144B, 16B aligned)
  __shared__ __bf16 Ks[64 * 72];   // [t][d]   (TDM pad reproduces 144B stride)
  __shared__ __bf16 Vt[64 * 72];   // [d][t]  (transposed -> B-frag rows)
  __shared__ __bf16 Ps[64 * 72];   // [q][t]  (C-layout -> A-layout bounce)
  const int tid = threadIdx.x;
  const int lane = tid & 31;
  const int wid = tid >> 5;
  const int mrow = lane & 15;
  const int abase = (lane < 16) ? 0 : 8;
  const int klo = (lane < 16) ? 0 : 16;
  const int h8 = (lane >= 16) ? 8 : 0;
  const int ncol = lane & 15;
  const int qt = blockIdx.x;
  const int hh = blockIdx.y;
  const int bb = blockIdx.z;
  const int qoff = hh * kD;
  const int koff = kE + hh * kD;
  const int voff = 2 * kE + hh * kD;
  const int tok0 = bb * kT;

  // stage Q tile (async; first wait_async0 below covers it)
#pragma unroll
  for (int c = 0; c < 2; ++c) {
    int chunk = tid + c * 128;
    int row = chunk >> 2, cc = chunk & 3;
    async_ld16((unsigned)(size_t)&Qs[row * 72 + cc * 16],
               &qkv[(size_t)(tok0 + qt * 64 + row) * (3 * kE) + qoff + cc * 16]);
  }

  const f32x8 vzero = {0.f, 0.f, 0.f, 0.f, 0.f, 0.f, 0.f, 0.f};
  f32x8 O[4];
#pragma unroll
  for (int dt = 0; dt < 4; ++dt) O[dt] = vzero;
  float mrun[8], lrun[8];
#pragma unroll
  for (int r = 0; r < 8; ++r) { mrun[r] = -INFINITY; lrun[r] = 0.f; }

  for (int j = 0; j <= qt; ++j) {
    __syncthreads();   // previous iter's K/V/P consumers done
    // K tile [t][d]: one TDM descriptor from wave 0 (fallback: async copies)
#if defined(HAVE_TDM)
    if (tid < 32)
      tdm_load_tile_64x64_bf16((unsigned)(size_t)&Ks[0],
          &qkv[(size_t)(tok0 + j * 64) * (3 * kE) + koff], 3 * kE);
#else
#pragma unroll
    for (int c = 0; c < 2; ++c) {
      int chunk = tid + c * 128;
      int row = chunk >> 2, cc = chunk & 3;
      async_ld16((unsigned)(size_t)&Ks[row * 72 + cc * 16],
                 &qkv[(size_t)(tok0 + j * 64 + row) * (3 * kE) + koff + cc * 16]);
    }
#endif
    // V tile transposed to [d][t] (manual: needs transpose)
#pragma unroll
    for (int g = 0; g < 8; ++g) {
      int gid = tid + g * 128;           // 0..1023
      int trow = gid >> 4, dc = gid & 15;
      uint2 raw = *(const uint2*)&qkv[(size_t)(tok0 + j * 64 + trow) * (3 * kE) + voff + dc * 4];
      const __bf16* ep = (const __bf16*)&raw;
#pragma unroll
      for (int i = 0; i < 4; ++i) Vt[(dc * 4 + i) * 72 + trow] = ep[i];
    }
    wait_async0();
#if defined(HAVE_TDM)
    if (tid < 32) __builtin_amdgcn_s_wait_tensorcnt(0);
#endif
    __syncthreads();

    // S = Q K^T (scaled); Q fragments re-read from resident Qs (no spills)
    f32x8 S[4];
    {
      bf16x16 aQ0 = frag_a(&Qs[(wid * 16 + mrow) * 72], abase);
      bf16x16 aQ1 = frag_a(&Qs[(wid * 16 + mrow) * 72 + 32], abase);
#pragma unroll
      for (int nt = 0; nt < 4; ++nt) {
        bf16x16 bk0 = frag_b(&Ks[(nt * 16 + mrow) * 72], klo);
        bf16x16 bk1 = frag_b(&Ks[(nt * 16 + mrow) * 72 + 32], klo);
        S[nt] = wmma_bf16(aQ0, bk0, vzero);
        S[nt] = wmma_bf16(aQ1, bk1, S[nt]);
      }
    }
#pragma unroll
    for (int nt = 0; nt < 4; ++nt)
#pragma unroll
      for (int r = 0; r < 8; ++r) {
        float s = S[nt][r] * 0.125f;     // 1/sqrt(64)
        int tg = j * 64 + nt * 16 + ncol;
        int qg = qt * 64 + wid * 16 + r + h8;
        S[nt][r] = (tg > qg) ? -INFINITY : s;
      }
    // online softmax (per row, across 16-lane half + 4 n-tiles)
#pragma unroll
    for (int r = 0; r < 8; ++r) {
      float mx = S[0][r];
#pragma unroll
      for (int nt = 1; nt < 4; ++nt) mx = fmaxf(mx, S[nt][r]);
#pragma unroll
      for (int msk = 1; msk < 16; msk <<= 1) mx = fmaxf(mx, __shfl_xor(mx, msk, 32));
      float mnew = fmaxf(mrun[r], mx);
      float alpha = __expf(mrun[r] - mnew);
      float sum = 0.f;
#pragma unroll
      for (int nt = 0; nt < 4; ++nt) {
        float pv = __expf(S[nt][r] - mnew);
        S[nt][r] = pv;
        sum += pv;
      }
#pragma unroll
      for (int msk = 1; msk < 16; msk <<= 1) sum += __shfl_xor(sum, msk, 32);
      lrun[r] = lrun[r] * alpha + sum;
      mrun[r] = mnew;
#pragma unroll
      for (int dt = 0; dt < 4; ++dt) O[dt][r] *= alpha;
    }
    // C-layout P -> bf16 LDS (own rows only; same-wave DS ordering, no barrier)
#pragma unroll
    for (int nt = 0; nt < 4; ++nt)
#pragma unroll
      for (int r = 0; r < 8; ++r)
        Ps[(wid * 16 + r + h8) * 72 + nt * 16 + ncol] = (__bf16)S[nt][r];
    // O += P V
#pragma unroll
    for (int ks = 0; ks < 2; ++ks) {
      bf16x16 ap = frag_a(&Ps[(wid * 16 + mrow) * 72 + ks * 32], abase);
#pragma unroll
      for (int dt = 0; dt < 4; ++dt) {
        bf16x16 bv = frag_b(&Vt[(dt * 16 + mrow) * 72 + ks * 32], klo);
        O[dt] = wmma_bf16(ap, bv, O[dt]);
      }
    }
  }
  // normalize + write y (interleaved back to [tok][h*64+d])
#pragma unroll
  for (int dt = 0; dt < 4; ++dt)
#pragma unroll
    for (int r = 0; r < 8; ++r) {
      float o = O[dt][r] / lrun[r];
      int tok = tok0 + qt * 64 + wid * 16 + r + h8;
      y[(size_t)tok * kE + hh * kD + dt * 16 + ncol] = (__bf16)o;
    }
}

// ---------------------------------------------------------------------------
// Host orchestration (graph-capture safe: only kernel launches on `stream`)
// ---------------------------------------------------------------------------
extern "C" void kernel_launch(void* const* d_in, const int* in_sizes, int n_in,
                              void* d_out, int out_size, void* d_ws, size_t ws_size,
                              hipStream_t stream) {
  (void)in_sizes; (void)n_in; (void)out_size; (void)ws_size;
  const int*   ids   = (const int*)d_in[0];
  const float* wte   = (const float*)d_in[1];
  const float* wpe   = (const float*)d_in[2];
  const float* ln1w  = (const float*)d_in[3];
  const float* ln1b  = (const float*)d_in[4];
  const float* wqkv  = (const float*)d_in[5];
  const float* bqkv  = (const float*)d_in[6];
  const float* wproj = (const float*)d_in[7];
  const float* bproj = (const float*)d_in[8];
  const float* ln2w  = (const float*)d_in[9];
  const float* ln2b  = (const float*)d_in[10];
  const float* wfc   = (const float*)d_in[11];
  const float* bfc   = (const float*)d_in[12];
  const float* wmp   = (const float*)d_in[13];
  const float* bmp   = (const float*)d_in[14];
  const float* lnfw  = (const float*)d_in[15];
  const float* lnfb  = (const float*)d_in[16];
  const float* lmw   = (const float*)d_in[17];
  float* out = (float*)d_out;

  char* p = (char*)d_ws;
  float*  x    = (float*)p;  p += (size_t)kTok * kE * 4;        // fp32 residual
  __bf16* h    = (__bf16*)p; p += (size_t)kTok * kE * 2;        // LN output
  __bf16* qkvb = (__bf16*)p; p += (size_t)kTok * 3 * kE * 2;    // QKV
  __bf16* yb   = (__bf16*)p; p += (size_t)kTok * kE * 2;        // attn out
  __bf16* fcb  = (__bf16*)p; p += (size_t)kTok * 4 * kE * 2;    // FC/GELU out

  embed_kernel<<<kTok, 256, 0, stream>>>(ids, wte, wpe, x);
  for (int l = 0; l < kL; ++l) {
    ln_kernel<<<kTok, 256, 0, stream>>>(x, ln1w + (size_t)l * kE, ln1b + (size_t)l * kE, h);
    gemm_kernel<0><<<dim3(32, 36), 128, 0, stream>>>(
        h, wqkv + (size_t)l * kE * 3 * kE, bqkv + (size_t)l * 3 * kE,
        nullptr, nullptr, qkvb, kE, 3 * kE);
    attn_kernel<<<dim3(kT / 64, kH, kB), 128, 0, stream>>>(qkvb, yb);
    gemm_kernel<1><<<dim3(32, 12), 128, 0, stream>>>(
        yb, wproj + (size_t)l * kE * kE, bproj + (size_t)l * kE,
        x, x, nullptr, kE, kE);
    ln_kernel<<<kTok, 256, 0, stream>>>(x, ln2w + (size_t)l * kE, ln2b + (size_t)l * kE, h);
    gemm_kernel<2><<<dim3(32, 48), 128, 0, stream>>>(
        h, wfc + (size_t)l * kE * 4 * kE, bfc + (size_t)l * 4 * kE,
        nullptr, nullptr, fcb, kE, 4 * kE);
    gemm_kernel<1><<<dim3(32, 12), 128, 0, stream>>>(
        fcb, wmp + (size_t)l * 4 * kE * kE, bmp + (size_t)l * kE,
        x, x, nullptr, 4 * kE, kE);
  }
  ln_kernel<<<kTok, 256, 0, stream>>>(x, lnfw, lnfb, h);
  gemm_kernel<3><<<dim3(32, (kV + 63) / 64), 128, 0, stream>>>(
      h, lmw, nullptr, nullptr, out, nullptr, kE, kV);
}
